// Get_Correlation_62148176773662
// MI455X (gfx1250) — matrix operand order repaired and checked by reading, not compiled, not run on hardware
//
#include <hip/hip_runtime.h>
#include <hip/hip_bf16.h>

// ---------------------------------------------------------------------------
// Problem constants (reference shapes: N=2, C=512, T=96, H=W=14, rc=32, nb=3)
// ---------------------------------------------------------------------------
#define N_DIM 2
#define C_DIM 512
#define T_DIM 96
#define HW_DIM 196
#define S_DIM (T_DIM * HW_DIM)      // 18816 positions per batch
#define B_DIM (N_DIM * T_DIM)       // 192 (n*T + t)
#define RC_DIM 32

typedef __attribute__((ext_vector_type(16))) __bf16 v16bf;
typedef __attribute__((ext_vector_type(8)))  float  v8f;
typedef __attribute__((ext_vector_type(4)))  unsigned u32x4;
typedef __attribute__((ext_vector_type(8)))  int      i32x8;
typedef __attribute__((ext_vector_type(4)))  int      i32x4;

union BF16x16 { v16bf v; unsigned short u[16]; };

__device__ __forceinline__ unsigned short f2bf(float f) {
    union { float f; unsigned u; } x; x.f = f;
    unsigned r = x.u + 0x7FFFu + ((x.u >> 16) & 1u);   // round-to-nearest-even
    return (unsigned short)(r >> 16);
}

__device__ __forceinline__ float sigm_m05(float a) {
    return 1.f / (1.f + __expf(-a)) - 0.5f;
}

// Issue a 2D TDM tile load: tile_dim0 x tile_dim1 elements of 4B, row stride
// `stride` elements, from `gaddr` into LDS byte offset `lds_off`.
__device__ __forceinline__ void tdm_load_2d_f32(unsigned lds_off, const float* gaddr,
                                                unsigned tile0, unsigned tile1,
                                                unsigned dim0, unsigned dim1,
                                                unsigned stride) {
    unsigned long long ga = (unsigned long long)(size_t)gaddr;
    u32x4 g0;
    g0.x = 1u;                                            // count=1, user mode
    g0.y = lds_off;                                       // LDS byte address
    g0.z = (unsigned)(ga & 0xFFFFFFFFu);                  // global_addr[31:0]
    g0.w = (unsigned)((ga >> 32) & 0x01FFFFFFu) | (2u << 30); // ga[56:32] | type=2
    i32x8 g1;
    g1.s0 = (int)(2u << 16);                              // data_size = 4B
    g1.s1 = (int)((dim0 & 0xFFFFu) << 16);                // tensor_dim0[15:0]
    g1.s2 = (int)((dim0 >> 16) | ((dim1 & 0xFFFFu) << 16)); // dim0[31:16] | dim1[15:0]
    g1.s3 = (int)((dim1 >> 16) | (tile0 << 16));          // dim1[31:16] | tile_dim0
    g1.s4 = (int)tile1;                                   // tile_dim1 (tile_dim2=0)
    g1.s5 = (int)stride;                                  // tensor_dim0_stride[31:0]
    g1.s6 = 0; g1.s7 = 0;
    i32x4 z4 = {0, 0, 0, 0};
#if __clang_major__ >= 23
    i32x8 z8 = {0, 0, 0, 0, 0, 0, 0, 0};
    __builtin_amdgcn_tensor_load_to_lds(g0, g1, z4, z4, z8, 0);
#else
    __builtin_amdgcn_tensor_load_to_lds(g0, g1, z4, z4, 0);
#endif
}

// ---------------------------------------------------------------------------
// 1) qp = (q_w @ query + q_b) * C^-0.5 ; kq = k_w^T qp ; qbdot = qp . k_b
//    (collapses the entire kp GEMM into one 512-vector)
// ---------------------------------------------------------------------------
__global__ void qp_kernel(const float* __restrict__ query, const float* __restrict__ q_w,
                          const float* __restrict__ q_b, const float* __restrict__ k_w,
                          const float* __restrict__ k_b,
                          float* __restrict__ kq, float* __restrict__ qbdot) {
    __shared__ float qps[C_DIM];
    __shared__ float red[512];
    int tid = threadIdx.x;                      // 512 threads
    float a = 0.f;
    for (int i = 0; i < C_DIM; ++i) a += q_w[(size_t)tid * C_DIM + i] * query[i];
    a = (a + q_b[tid]) * 0.044194173824159216f; // 512^-0.5
    qps[tid] = a;
    __syncthreads();
    float s = 0.f;
    for (int c = 0; c < C_DIM; ++c) s += k_w[(size_t)c * C_DIM + tid] * qps[c];
    kq[tid] = s;
    red[tid] = qps[tid] * k_b[tid];
    __syncthreads();
    for (int st = 256; st > 0; st >>= 1) {
        if (tid < st) red[tid] += red[tid + st];
        __syncthreads();
    }
    if (tid == 0) qbdot[0] = red[0];
}

// ---------------------------------------------------------------------------
// 2) Per (n,t): logits = kq.x + qbdot -> softmax over 196 -> attention pooling
// ---------------------------------------------------------------------------
__global__ void attn_pool_kernel(const float* __restrict__ x,
                                 const float* __restrict__ kq,
                                 const float* __restrict__ qb,
                                 float* __restrict__ xmean, float* __restrict__ xmax,
                                 float* __restrict__ xw) {
    int b = blockIdx.x, n = b / T_DIM, t = b % T_DIM;
    const float* xb = x + ((size_t)n * C_DIM) * S_DIM + (size_t)t * HW_DIM;
    __shared__ float kqs[C_DIM];
    __shared__ float red[256];
    __shared__ float attn[HW_DIM];
    int tid = threadIdx.x;                      // 256 threads
    for (int c = tid; c < C_DIM; c += 256) kqs[c] = kq[c];
    __syncthreads();

    float acc = 0.f;
    if (tid < HW_DIM) {
        for (int c = 0; c < C_DIM; ++c) acc += kqs[c] * xb[(size_t)c * S_DIM + tid];
        acc += qb[0];
    }
    for (int cc = 0; cc < 2; ++cc) {            // mean / max, 2 channels per thread
        int c = tid + cc * 256;
        const float* row = xb + (size_t)c * S_DIM;
        float s = 0.f, mx = -3.4e38f;
        for (int i = 0; i < HW_DIM; ++i) { float v = row[i]; s += v; mx = fmaxf(mx, v); }
        xmean[(size_t)b * C_DIM + c] = s * (1.0f / HW_DIM);
        xmax [(size_t)b * C_DIM + c] = mx;
    }
    red[tid] = (tid < HW_DIM) ? acc : -3.4e38f;
    __syncthreads();
    for (int s = 128; s > 0; s >>= 1) { if (tid < s) red[tid] = fmaxf(red[tid], red[tid + s]); __syncthreads(); }
    float mval = red[0];
    __syncthreads();
    float e = (tid < HW_DIM) ? __expf(acc - mval) : 0.f;
    red[tid] = e;
    __syncthreads();
    for (int s = 128; s > 0; s >>= 1) { if (tid < s) red[tid] += red[tid + s]; __syncthreads(); }
    float inv = 1.f / red[0];
    if (tid < HW_DIM) attn[tid] = e * inv;
    __syncthreads();
    for (int cc = 0; cc < 2; ++cc) {
        int c = tid + cc * 256;
        const float* row = xb + (size_t)c * S_DIM;
        float s = 0.f;
        for (int i = 0; i < HW_DIM; ++i) s += attn[i] * row[i];
        xw[(size_t)b * C_DIM + c] = s;
    }
}

// ---------------------------------------------------------------------------
// Generic small GEMM on WMMA bf16 (compile-time variants, no runtime flags):
//   WT==0: D[m,n] = sum_k A[m,k]*W[n*K+k] (+bias)   (out = A @ W^T)
//   WT==1: D[m,n] = sum_k A[m,k]*W[k*N+n] (+bias)   (out = A @ W)
// One wave per 16x16 tile, K stepped by 32.
// ---------------------------------------------------------------------------
template <int WT, bool HASB>
__global__ void gemm_wmma(const float* __restrict__ A, const float* __restrict__ W,
                          const float* __restrict__ bias, float* __restrict__ D,
                          int N, int K) {
    int ntn = N >> 4;
    int m0 = (blockIdx.x / ntn) << 4;
    int n0 = (blockIdx.x % ntn) << 4;
    int lane = threadIdx.x, lm = lane & 15, half = lane >> 4;
    v8f acc = {};
    for (int kk = 0; kk < K; kk += 32) {
        BF16x16 a, b;
#pragma unroll
        for (int e = 0; e < 16; ++e) {
            int ka = e + (half ? 16 : 0);
            a.u[e] = f2bf(A[(size_t)(m0 + lm) * K + kk + ka]);
            int kb = e + (half ? 16 : 0);
            float wv;
            if (WT) wv = W[(size_t)(kk + kb) * N + n0 + lm];
            else    wv = W[(size_t)(n0 + lm) * K + kk + kb];
            b.u[e] = f2bf(wv);
        }
        acc = __builtin_amdgcn_wmma_f32_16x16x32_bf16(false, a.v, false, b.v,
                                                      (short)0, acc, false, false);
    }
    float bv = HASB ? bias[n0 + lm] : 0.f;
#pragma unroll
    for (int v = 0; v < 8; ++v) {
        int m = m0 + v + (half ? 8 : 0);
        D[(size_t)m * N + n0 + lm] = acc[v] + bv;
    }
}

// ---------------------------------------------------------------------------
// 3) xq = w4[0]*mean + w4[1]*max + w4[2]*att_out   (elementwise, B*C)
// ---------------------------------------------------------------------------
__global__ void xq_kernel(const float* __restrict__ xmean, const float* __restrict__ xmax,
                          const float* __restrict__ attout, const float* __restrict__ w4,
                          float* __restrict__ xq) {
    int i = blockIdx.x * 256 + threadIdx.x;
    if (i < B_DIM * C_DIM)
        xq[i] = xmean[i] * w4[0] + xmax[i] * w4[1] + attout[i] * w4[2];
}

// ---------------------------------------------------------------------------
// 4) Gate: slice-centric. Slice t' serves queries t = t' - off_j,
//    off_j in {-3,-2,-1,1,2,3}. g[b,j,hw] = sigmoid(w2[j]*(xqT[b].x_col)) - 0.5
// ---------------------------------------------------------------------------
__global__ void gate_kernel(const float* __restrict__ x, const float* __restrict__ xqT,
                            const float* __restrict__ w2, float* __restrict__ g) {
    int bq = blockIdx.x, n = bq / T_DIM, tp = bq % T_DIM;
    __shared__ float qrows[6 * C_DIM];
    __shared__ float w2s[6];
    int tid = threadIdx.x;                      // 256
    if (tid < 6) w2s[tid] = w2[tid];
    for (int j = 0; j < 6; ++j) {
        int off = (j < 3) ? j - 3 : j - 2;
        int tq = tp - off;
        bool ok = (tq >= 0 && tq < T_DIM);
        for (int c = tid; c < C_DIM; c += 256)
            qrows[j * C_DIM + c] = ok ? xqT[(size_t)(n * T_DIM + tq) * C_DIM + c] : 0.f;
    }
    __syncthreads();
    if (tid < HW_DIM) {
        const float* xs = x + ((size_t)n * C_DIM) * S_DIM + (size_t)tp * HW_DIM + tid;
        float d0 = 0, d1 = 0, d2 = 0, d3 = 0, d4 = 0, d5 = 0;
        for (int c = 0; c < C_DIM; ++c) {
            __builtin_prefetch(xs + (size_t)(c + 16) * S_DIM, 0, 0);
            float v = xs[(size_t)c * S_DIM];
            d0 += qrows[0 * C_DIM + c] * v; d1 += qrows[1 * C_DIM + c] * v;
            d2 += qrows[2 * C_DIM + c] * v; d3 += qrows[3 * C_DIM + c] * v;
            d4 += qrows[4 * C_DIM + c] * v; d5 += qrows[5 * C_DIM + c] * v;
        }
        float d[6] = { d0, d1, d2, d3, d4, d5 };
        for (int j = 0; j < 6; ++j) {
            int off = (j < 3) ? j - 3 : j - 2;
            int tq = tp - off;
            if (tq >= 0 && tq < T_DIM)
                g[((size_t)(n * T_DIM + tq) * 6 + j) * HW_DIM + tid] = sigm_m05(w2s[j] * d[j]);
        }
    }
}

// ---------------------------------------------------------------------------
// 5) pooled[b,c] = sum_j w2[j] * sum_hw x[n,c,t+off_j,hw] * g[b,j,hw]
// ---------------------------------------------------------------------------
__global__ void pooled_kernel(const float* __restrict__ x, const float* __restrict__ g,
                              const float* __restrict__ w2, float* __restrict__ pooled) {
    int b = blockIdx.x, n = b / T_DIM, t = b % T_DIM;
    __shared__ float gs[6 * HW_DIM];
    __shared__ int toff[6];
    int tid = threadIdx.x;                      // 256
    if (tid < 6) {
        int off = (tid < 3) ? tid - 3 : tid - 2;
        int ts = t + off;
        toff[tid] = (ts >= 0 && ts < T_DIM) ? ts : -1;
    }
    __syncthreads();
    for (int j = 0; j < 6; ++j) {
        float wv = w2[j];
        for (int i = tid; i < HW_DIM; i += 256)
            gs[j * HW_DIM + i] = (toff[j] >= 0) ? wv * g[((size_t)b * 6 + j) * HW_DIM + i] : 0.f;
    }
    __syncthreads();
    for (int cc = 0; cc < 2; ++cc) {
        int c = tid + cc * 256;
        float acc = 0.f;
        for (int j = 0; j < 6; ++j) {
            int ts = toff[j];
            if (ts < 0) continue;
            const float* row = x + ((size_t)n * C_DIM + c) * S_DIM + (size_t)ts * HW_DIM;
            const float* gr = gs + j * HW_DIM;
            for (int i = 0; i < HW_DIM; ++i) acc += gr[i] * row[i];
        }
        pooled[(size_t)b * C_DIM + c] = acc;
    }
}

// ---------------------------------------------------------------------------
// 6) x_down = down_conv_w (32x512) @ x via WMMA. The 32x32 f32 x-tile is moved
//    into LDS by the Tensor Data Mover (TDM); wave 0 issues the descriptor,
//    waits on TENSORcnt, and a workgroup barrier publishes the tile.
//    Block: 128 threads (4 waves) -> 32 rows x 32 positions. K loop 512/32.
// ---------------------------------------------------------------------------
__global__ void xdown_kernel(const float* __restrict__ x, const float* __restrict__ dw,
                             float* __restrict__ xdown) {
    int blk = blockIdx.x;
    int n = blk / (S_DIM / 32);
    int p0 = (blk % (S_DIM / 32)) * 32;
    __shared__ float xtf[32][32];               // [k][p], filled by TDM
    __shared__ unsigned short wt[32][32];       // [m][k]
    int tid = threadIdx.x;
    int wave = tid >> 5, lane = tid & 31, lm = lane & 15, half = lane >> 4;
    int m0 = (wave & 1) * 16, pb = (wave >> 1) * 16;
    unsigned lds_off = (unsigned)(size_t)&xtf[0][0];   // LDS aperture: low 32 bits
    v8f acc = {};
    for (int kk = 0; kk < C_DIM; kk += 32) {
        __syncthreads();                        // previous tile fully consumed
        if (tid < 32) {                         // one TDM op per block
            tdm_load_2d_f32(lds_off,
                            x + ((size_t)n * C_DIM + kk) * S_DIM + p0,
                            /*tile0=*/32, /*tile1=*/32,
                            /*dim0=*/S_DIM, /*dim1=*/C_DIM, /*stride=*/S_DIM);
            __builtin_amdgcn_s_wait_tensorcnt(0);
        }
        for (int i = tid; i < 1024; i += 128) { // weight tile (plain VMEM)
            int m = i & 31, k = i >> 5;
            wt[m][k] = f2bf(dw[(size_t)m * C_DIM + kk + k]);
        }
        __syncthreads();                        // TDM tile + weights visible
        BF16x16 a, b;
#pragma unroll
        for (int e = 0; e < 16; ++e) {
            int ka = e + (half ? 16 : 0);
            a.u[e] = wt[m0 + lm][ka];
            int kb = e + (half ? 16 : 0);
            b.u[e] = f2bf(xtf[kb][pb + lm]);
        }
        acc = __builtin_amdgcn_wmma_f32_16x16x32_bf16(false, a.v, false, b.v,
                                                      (short)0, acc, false, false);
    }
#pragma unroll
    for (int v = 0; v < 8; ++v) {
        int m = m0 + v + (half ? 8 : 0);
        xdown[((size_t)n * RC_DIM + m) * S_DIM + p0 + pb + lm] = acc[v];
    }
}

// ---------------------------------------------------------------------------
// 7) Fused 3-branch dilated depthwise conv (9x3x3, dil 1/2/3) on rc=32.
// ---------------------------------------------------------------------------
__global__ void dwconv_kernel(const float* __restrict__ xd,
                              const float* __restrict__ w1, const float* __restrict__ b1,
                              const float* __restrict__ w2_, const float* __restrict__ b2,
                              const float* __restrict__ w3, const float* __restrict__ b3,
                              const float* __restrict__ wts, float* __restrict__ agg) {
    int blk = blockIdx.x;
    int t = blk % T_DIM, r = (blk / T_DIM) % RC_DIM, n = blk / (T_DIM * RC_DIM);
    __shared__ float ws[3][81];
    __shared__ float bsum;
    int tid = threadIdx.x;                      // 256
    if (tid < 81) {
        ws[0][tid] = w1 [(size_t)r * 81 + tid] * wts[0];
        ws[1][tid] = w2_[(size_t)r * 81 + tid] * wts[1];
        ws[2][tid] = w3 [(size_t)r * 81 + tid] * wts[2];
    }
    if (tid == 0) bsum = b1[r] * wts[0] + b2[r] * wts[1] + b3[r] * wts[2];
    __syncthreads();
    if (tid >= HW_DIM) return;
    int h = tid / 14, w = tid % 14;
    const float* base = xd + ((size_t)n * RC_DIM + r) * S_DIM;
    float acc = bsum;
    for (int br = 0; br < 3; ++br) {
        int dil = br + 1;
        for (int kt = 0; kt < 9; ++kt) {
            int ts = t + kt - 4;
            if (ts < 0 || ts >= T_DIM) continue;
            const float* sl = base + (size_t)ts * HW_DIM;
            for (int kh = 0; kh < 3; ++kh) {
                int hh = h + (kh - 1) * dil;
                if (hh < 0 || hh >= 14) continue;
                for (int kw = 0; kw < 3; ++kw) {
                    int ww = w + (kw - 1) * dil;
                    if (ww < 0 || ww >= 14) continue;
                    acc += ws[br][kt * 9 + kh * 3 + kw] * sl[hh * 14 + ww];
                }
            }
        }
    }
    agg[((size_t)n * RC_DIM + r) * S_DIM + (size_t)t * HW_DIM + tid] = acc;
}

// ---------------------------------------------------------------------------
// 8) Output: per 16-position tile, conv_back (512x32) @ agg tile via one
//    WMMA (K=32), then out = features * (sigmoid(.) - 0.5). 8 waves / 512 rows.
// ---------------------------------------------------------------------------
__global__ void output_kernel(const float* __restrict__ agg, const float* __restrict__ cw,
                              const float* __restrict__ feat, float* __restrict__ out) {
    int blk = blockIdx.x;
    int n = blk / (S_DIM / 16);
    int p0 = (blk % (S_DIM / 16)) * 16;
    __shared__ unsigned short at[32][16];       // [r][p]
    int tid = threadIdx.x;                      // 256
    for (int i = tid; i < 512; i += 256) {
        int r = i >> 4, p = i & 15;
        at[r][p] = f2bf(agg[((size_t)n * RC_DIM + r) * S_DIM + p0 + p]);
    }
    __syncthreads();
    int wave = tid >> 5, lane = tid & 31, lm = lane & 15, half = lane >> 4;
    int p = p0 + lm;
    int t = p / HW_DIM;
    for (int sub = 0; sub < 4; ++sub) {
        int c0 = (wave * 4 + sub) * 16;
        BF16x16 a, b;
#pragma unroll
        for (int e = 0; e < 16; ++e) {
            int ka = e + (half ? 16 : 0);
            a.u[e] = f2bf(cw[(size_t)(c0 + lm) * RC_DIM + ka]);
            int kb = e + (half ? 16 : 0);
            b.u[e] = at[kb][lm];
        }
        v8f acc = {};
        acc = __builtin_amdgcn_wmma_f32_16x16x32_bf16(false, a.v, false, b.v,
                                                      (short)0, acc, false, false);
#pragma unroll
        for (int v = 0; v < 8; ++v) {
            int c = c0 + v + (half ? 8 : 0);
            float gate = sigm_m05(acc[v]);
            out[((size_t)n * C_DIM + c) * S_DIM + p] =
                feat[(size_t)(n * T_DIM + t) * C_DIM + c] * gate;
        }
    }
}

// ---------------------------------------------------------------------------
// Host launcher
// ---------------------------------------------------------------------------
extern "C" void kernel_launch(void* const* d_in, const int* in_sizes, int n_in,
                              void* d_out, int out_size, void* d_ws, size_t ws_size,
                              hipStream_t stream) {
    (void)in_sizes; (void)n_in; (void)out_size; (void)ws_size;
    const float* x     = (const float*)d_in[0];
    const float* W2    = (const float*)d_in[1];   // down_conv2_w (C,C)
    const float* Wd    = (const float*)d_in[2];   // down_conv_w  (rc,C)
    const float* Wb    = (const float*)d_in[3];   // conv_back_w  (C,rc)
    const float* w2v   = (const float*)d_in[4];   // weights2 (6)
    const float* w4    = (const float*)d_in[5];   // weights4 (3)
    const float* wts   = (const float*)d_in[6];   // weights  (3)
    const float* query = (const float*)d_in[7];
    const float* q_w   = (const float*)d_in[8];
    const float* q_b   = (const float*)d_in[9];
    const float* k_w   = (const float*)d_in[10];
    const float* k_b   = (const float*)d_in[11];
    const float* v_w   = (const float*)d_in[12];
    const float* v_b   = (const float*)d_in[13];
    const float* c_w   = (const float*)d_in[14];
    const float* c_b   = (const float*)d_in[15];
    const float* sa1w  = (const float*)d_in[16];
    const float* sa1b  = (const float*)d_in[17];
    const float* sa2w  = (const float*)d_in[18];
    const float* sa2b  = (const float*)d_in[19];
    const float* sa3w  = (const float*)d_in[20];
    const float* sa3b  = (const float*)d_in[21];

    float* ws = (float*)d_ws;
    size_t off = 0;
    auto alloc = [&](size_t nelem) {
        float* p = ws + off;
        off += (nelem + 63) & ~(size_t)63;
        return p;
    };
    float* kq     = alloc(C_DIM);
    float* qbdot  = alloc(1);
    float* xmean  = alloc((size_t)B_DIM * C_DIM);
    float* xmax   = alloc((size_t)B_DIM * C_DIM);
    float* xw     = alloc((size_t)B_DIM * C_DIM);
    float* vp     = alloc((size_t)B_DIM * C_DIM);
    float* attout = alloc((size_t)B_DIM * C_DIM);
    float* xq     = alloc((size_t)B_DIM * C_DIM);
    float* xqT    = alloc((size_t)B_DIM * C_DIM);
    float* pooled = alloc((size_t)B_DIM * C_DIM);
    float* feats  = alloc((size_t)B_DIM * C_DIM);
    float* g      = alloc((size_t)B_DIM * 6 * HW_DIM);
    float* xdown  = alloc((size_t)N_DIM * RC_DIM * S_DIM);
    float* agg    = alloc((size_t)N_DIM * RC_DIM * S_DIM);

    int gemm_grid = (B_DIM / 16) * (C_DIM / 16);   // 12*32 = 384 waves

    qp_kernel<<<1, 512, 0, stream>>>(query, q_w, q_b, k_w, k_b, kq, qbdot);
    attn_pool_kernel<<<B_DIM, 256, 0, stream>>>(x, kq, qbdot, xmean, xmax, xw);
    gemm_wmma<0, true ><<<gemm_grid, 32, 0, stream>>>(xw, v_w, v_b, vp, C_DIM, C_DIM);
    gemm_wmma<0, true ><<<gemm_grid, 32, 0, stream>>>(vp, c_w, c_b, attout, C_DIM, C_DIM);
    xq_kernel<<<(B_DIM * C_DIM + 255) / 256, 256, 0, stream>>>(xmean, xmax, attout, w4, xq);
    gemm_wmma<1, false><<<gemm_grid, 32, 0, stream>>>(xq, W2, nullptr, xqT, C_DIM, C_DIM);
    gate_kernel<<<B_DIM, 256, 0, stream>>>(x, xqT, w2v, g);
    pooled_kernel<<<B_DIM, 256, 0, stream>>>(x, g, w2v, pooled);
    gemm_wmma<0, false><<<gemm_grid, 32, 0, stream>>>(pooled, W2, nullptr, feats, C_DIM, C_DIM);
    xdown_kernel<<<N_DIM * (S_DIM / 32), 128, 0, stream>>>(x, Wd, xdown);
    dwconv_kernel<<<N_DIM * RC_DIM * T_DIM, 256, 0, stream>>>(xdown, sa1w, sa1b, sa2w, sa2b,
                                                              sa3w, sa3b, wts, agg);
    output_kernel<<<N_DIM * (S_DIM / 16), 256, 0, stream>>>(agg, Wb, feats, (float*)d_out);
}
